// MixtureOfExperts_24120536335016
// MI455X (gfx1250) — matrix operand order repaired and checked by reading, not compiled, not run on hardware
//
#include <hip/hip_runtime.h>
#include <hip/hip_bf16.h>
#include <math.h>

#define D_DIM  1024
#define F_DIM  4096
#define E_NUM  8
#define T_TOK  8192      // B*S
#define TILE_M 16
#define TILES_PER_E 512  // T_TOK / TILE_M

typedef __attribute__((ext_vector_type(16))) __bf16 bf16x16;
typedef __attribute__((ext_vector_type(8)))  __bf16 bf16x8;
typedef __attribute__((ext_vector_type(4)))  __bf16 bf16x4;
typedef __attribute__((ext_vector_type(8)))  float  f32x8;
typedef __attribute__((ext_vector_type(4)))  float  f32x4;

// Build a 16xbf16 A/B fragment from two contiguous 16-byte runs
// (matches CDNA5 §7.12.2: per-lane VGPR0-3 then VGPR4-7 K runs).
static __device__ __forceinline__ bf16x16 frag2(const __bf16* p0, const __bf16* p1) {
    union { bf16x16 f; bf16x8 h[2]; } u;
    u.h[0] = *(const bf16x8*)p0;
    u.h[1] = *(const bf16x8*)p1;
    return u.f;
}

// ---------------- fp32 -> bf16 conversion (x4 vectorized) ----------------
__global__ void cvt_bf16_kernel(const float* __restrict__ src,
                                __bf16* __restrict__ dst, long n4) {
    long i      = (long)blockIdx.x * blockDim.x + threadIdx.x;
    long stride = (long)gridDim.x * blockDim.x;
    const f32x4* s4 = (const f32x4*)src;
    bf16x4*      d4 = (bf16x4*)dst;
    for (; i < n4; i += stride) {
        f32x4 v = s4[i];
        d4[i] = __builtin_convertvector(v, bf16x4);
    }
}

// ---------------- router: logits, top-2 softmax, expert gather lists ----------------
__global__ void router_kernel(const float* __restrict__ x,
                              const float* __restrict__ rw,
                              float* __restrict__ gate,
                              int* __restrict__ counts,
                              int* __restrict__ lists) {
    const int lane = threadIdx.x & 31;
    const int wv   = threadIdx.x >> 5;
    const int t    = blockIdx.x * 8 + wv;
    if (t >= T_TOK) return;

    float acc[E_NUM];
#pragma unroll
    for (int e = 0; e < E_NUM; ++e) acc[e] = 0.0f;

    const float* xr = x + (size_t)t * D_DIM;
    for (int d = lane; d < D_DIM; d += 32) {
        float xv = xr[d];
#pragma unroll
        for (int e = 0; e < E_NUM; ++e) acc[e] += xv * rw[e * D_DIM + d];
    }
#pragma unroll
    for (int e = 0; e < E_NUM; ++e) {
        for (int off = 16; off > 0; off >>= 1)
            acc[e] += __shfl_xor(acc[e], off, 32);
    }
    if (lane == 0) {
        int   i1 = 0;   float v1 = acc[0];
#pragma unroll
        for (int e = 1; e < E_NUM; ++e) if (acc[e] > v1) { v1 = acc[e]; i1 = e; }
        int   i2 = -1;  float v2 = -3.4e38f;
#pragma unroll
        for (int e = 0; e < E_NUM; ++e) if (e != i1 && acc[e] > v2) { v2 = acc[e]; i2 = e; }
        // softmax over {v1, v2}, v1 >= v2
        float ew  = expf(v2 - v1);
        float inv = 1.0f / (1.0f + ew);
        float p1  = inv, p2 = ew * inv;
#pragma unroll
        for (int e = 0; e < E_NUM; ++e)
            gate[t * E_NUM + e] = (e == i1) ? p1 : ((e == i2) ? p2 : 0.0f);
        int pos1 = atomicAdd(&counts[i1], 1);
        lists[i1 * T_TOK + pos1] = t;
        int pos2 = atomicAdd(&counts[i2], 1);
        lists[i2 * T_TOK + pos2] = t;
    }
}

// ---------------- fused expert FFN: gelu(x@w1^T+b1)@w2^T+b2, gated scatter-add ----------------
__global__ __launch_bounds__(256)
void moe_ffn_kernel(const __bf16* __restrict__ xb,
                    const __bf16* __restrict__ w1b,
                    const __bf16* __restrict__ w2b,
                    const float* __restrict__ b1,
                    const float* __restrict__ b2,
                    const float* __restrict__ gate,
                    const int* __restrict__ counts,
                    const int* __restrict__ lists,
                    float* __restrict__ out) {
    __shared__ __bf16 h_lds[TILE_M][F_DIM + 8];   // padded stride: bank-conflict-free b128
    __shared__ int    tok_lds[TILE_M];
    __shared__ float  gate_lds[TILE_M];

    const int e    = blockIdx.x >> 9;       // / TILES_PER_E
    const int tile = blockIdx.x & 511;
    const int cnt  = counts[e];
    const int base = tile * TILE_M;
    if (base >= cnt) return;                // uniform per block

    const int lane = threadIdx.x & 31;
    const int wv   = threadIdx.x >> 5;      // 8 waves
    const int n16  = lane & 15;
    const int hv   = lane >> 4;

    if (threadIdx.x < TILE_M) {
        int idx = base + (int)threadIdx.x;
        if (idx < cnt) {
            int t = lists[e * T_TOK + idx];
            tok_lds[threadIdx.x]  = t;
            gate_lds[threadIdx.x] = gate[t * E_NUM + e];
        } else {
            tok_lds[threadIdx.x]  = 0;      // padded slot: gate 0 => contributes nothing
            gate_lds[threadIdx.x] = 0.0f;
        }
    }
    __syncthreads();

    // A-fragment row for this lane is m = n16 (lanes 0-15 and 16-31 share rows, differ in K)
    const __bf16* xrow = xb + (size_t)tok_lds[n16] * D_DIM;
    const __bf16* w1e  = w1b + (size_t)e * F_DIM * D_DIM;

    // ===== GEMM1: h[16 x 4096] = gelu(x_tile @ w1^T + b1) =====
    // wave wv owns columns [wv*512, wv*512+512); 8 groups of 4 n-tiles
    for (int ng = 0; ng < 8; ++ng) {
        const int nbase0 = wv * 512 + ng * 64;
        f32x8 acc[4] = {};
        for (int k0 = 0; k0 < D_DIM; k0 += 32) {
            bf16x16 a = frag2(xrow + k0 + hv * 8, xrow + k0 + 16 + hv * 8);
#pragma unroll
            for (int u = 0; u < 4; ++u) {
                const int ncol = nbase0 + u * 16 + n16;
                const __bf16* bp = w1e + (size_t)ncol * D_DIM + k0 + hv * 8;
                bf16x16 b = frag2(bp, bp + 16);
                acc[u] = __builtin_amdgcn_wmma_f32_16x16x32_bf16(
                    false, a, false, b, (short)0, acc[u], false, false);
            }
        }
#pragma unroll
        for (int u = 0; u < 4; ++u) {
            const int   ncol = nbase0 + u * 16 + n16;
            const float bias = b1[e * F_DIM + ncol];
#pragma unroll
            for (int j = 0; j < 8; ++j) {
                const int row = j + hv * 8;   // C/D layout: M = vgpr + 8*(lane>>4)
                float v = acc[u][j] + bias;
                float g = 0.5f * v * (1.0f + erff(v * 0.70710678118654752f)); // exact gelu
                h_lds[row][ncol] = (__bf16)g;
            }
        }
    }
    __syncthreads();

    // ===== GEMM2: y[16 x 1024] = h @ w2^T + b2, scaled by gate, scatter-add =====
    const __bf16* w2e = w2b + (size_t)e * (size_t)D_DIM * F_DIM;
    for (int ng = 0; ng < 2; ++ng) {
        const int nbase0 = wv * 128 + ng * 64;
        f32x8 acc[4] = {};
        for (int k0 = 0; k0 < F_DIM; k0 += 32) {
            bf16x16 a = frag2(&h_lds[n16][k0 + hv * 8], &h_lds[n16][k0 + 16 + hv * 8]);
#pragma unroll
            for (int u = 0; u < 4; ++u) {
                const int ncol = nbase0 + u * 16 + n16;
                const __bf16* bp = w2e + (size_t)ncol * F_DIM + k0 + hv * 8;
                bf16x16 b = frag2(bp, bp + 16);
                acc[u] = __builtin_amdgcn_wmma_f32_16x16x32_bf16(
                    false, a, false, b, (short)0, acc[u], false, false);
            }
        }
#pragma unroll
        for (int u = 0; u < 4; ++u) {
            const int   ncol = nbase0 + u * 16 + n16;
            const float bias = b2[e * D_DIM + ncol];
#pragma unroll
            for (int j = 0; j < 8; ++j) {
                const int row = j + hv * 8;
                float val = (acc[u][j] + bias) * gate_lds[row];
                atomicAdd(out + (size_t)tok_lds[row] * D_DIM + ncol, val);
            }
        }
    }
}

extern "C" void kernel_launch(void* const* d_in, const int* in_sizes, int n_in,
                              void* d_out, int out_size, void* d_ws, size_t ws_size,
                              hipStream_t stream) {
    const float* x  = (const float*)d_in[0];   // [T, D]
    const float* rw = (const float*)d_in[1];   // [E, D]
    const float* w1 = (const float*)d_in[2];   // [E, F, D]
    const float* b1 = (const float*)d_in[3];   // [E, F]
    const float* w2 = (const float*)d_in[4];   // [E, D, F]
    const float* b2 = (const float*)d_in[5];   // [E, D]
    float* out = (float*)d_out;                // [T, D]

    char* ws = (char*)d_ws;
    auto take = [&](size_t bytes) -> void* {
        void* p = (void*)ws;
        ws += (bytes + 255) & ~(size_t)255;
        return p;
    };
    __bf16* xb   = (__bf16*)take((size_t)T_TOK * D_DIM * 2);
    __bf16* w1b  = (__bf16*)take((size_t)E_NUM * F_DIM * D_DIM * 2);
    __bf16* w2b  = (__bf16*)take((size_t)E_NUM * F_DIM * D_DIM * 2);
    float*  gate = (float*)take((size_t)T_TOK * E_NUM * 4);
    int*  counts = (int*)take(256);
    int*  lists  = (int*)take((size_t)E_NUM * T_TOK * 4);

    hipMemsetAsync(counts, 0, E_NUM * sizeof(int), stream);
    hipMemsetAsync(out, 0, (size_t)out_size * sizeof(float), stream);

    cvt_bf16_kernel<<<2048, 256, 0, stream>>>(x,  xb,  (long)T_TOK * D_DIM / 4);
    cvt_bf16_kernel<<<4096, 256, 0, stream>>>(w1, w1b, (long)E_NUM * F_DIM * D_DIM / 4);
    cvt_bf16_kernel<<<4096, 256, 0, stream>>>(w2, w2b, (long)E_NUM * F_DIM * D_DIM / 4);

    router_kernel<<<T_TOK / 8, 256, 0, stream>>>(x, rw, gate, counts, lists);

    moe_ffn_kernel<<<E_NUM * TILES_PER_E, 256, 0, stream>>>(
        xb, w1b, w2b, b1, b2, gate, counts, lists, out);
}
